// SimplifiedTransformerBlock_44135083934205
// MI455X (gfx1250) — matrix operand order repaired and checked by reading, not compiled, not run on hardware
//
#include <hip/hip_runtime.h>
#include <hip/hip_bf16.h>
#include <math.h>
#include <stdint.h>

// ---------------------------------------------------------------------------
// Problem constants (fixed by the reference setup_inputs)
// ---------------------------------------------------------------------------
#define BB   2
#define TT   2048
#define CC   1024
#define HH   16
#define DD   64          // head dim = CC/HH
#define NFF  4096        // 4*CC
#define MM   (BB*TT)     // 4096 rows

typedef __attribute__((ext_vector_type(16))) __bf16        v16bf;
typedef __attribute__((ext_vector_type(8)))  float         v8f;
typedef __attribute__((ext_vector_type(4)))  unsigned int  v4u;
typedef __attribute__((ext_vector_type(4)))  int           v4i;
typedef __attribute__((ext_vector_type(8)))  int           v8i;

#if defined(__has_builtin)
#  if __has_builtin(__builtin_amdgcn_tensor_load_to_lds)
#    define HAVE_TDM 1
#  else
#    define HAVE_TDM 0
#  endif
#else
#  define HAVE_TDM 0
#endif

#if defined(__has_include)
#  if __has_include(<hip/amd_detail/amd_gfx1250_TDM.h>)
#    define TDM_ARGS6 1
#  else
#    define TDM_ARGS6 0
#  endif
#else
#  define TDM_ARGS6 0
#endif

// ---------------------------------------------------------------------------
// TDM: issue one 2D tile load (rows x cols of bf16) global -> LDS (async,
// tracked by TENSORcnt). D# layout per CDNA5 ISA.
// pad codes: interval 3 -> every 16 dwords (32 bf16) / interval 4 -> every 32
// dwords (64 bf16); amount 3 -> 4 dwords (8 bf16) => LDS strides 40 / 72.
// ---------------------------------------------------------------------------
#if HAVE_TDM
__device__ __forceinline__ void tdm_load_2d(const void* gptr, uint32_t lds_off,
                                            uint32_t tile_rows, uint32_t tile_cols,
                                            uint64_t row_stride_elems,
                                            uint32_t tensor_rows,
                                            uint32_t pad_int_code,
                                            uint32_t pad_amt_code) {
    const uint64_t ga = (uint64_t)(uintptr_t)gptr;
    v4u g0;
    g0.x = 1u;                                   // count=1, user descriptor
    g0.y = lds_off;                              // LDS byte address
    g0.z = (uint32_t)ga;                         // global_addr[31:0]
    g0.w = (uint32_t)(ga >> 32) | (2u << 30);    // global_addr[56:32] | type=2

    const uint32_t td0 = 0x7FFFFFFFu;            // dim0 bound: no clipping
    const uint64_t st0 = row_stride_elems;       // dim0 stride (elements)
    v8i g1;
    g1[0] = (int)((1u << 16) | (1u << 20) |
                  (pad_int_code << 22) | (pad_amt_code << 25));
    g1[1] = (int)((td0 & 0xFFFFu) << 16);                         // dim0 lo16
    g1[2] = (int)((td0 >> 16) | ((tensor_rows & 0xFFFFu) << 16)); // dim0 hi/dim1 lo
    g1[3] = (int)((tensor_rows >> 16) | (tile_cols << 16));       // dim1 hi/tile0
    g1[4] = (int)(tile_rows & 0xFFFFu);                           // tile1, tile2=0
    g1[5] = (int)(uint32_t)st0;                                   // stride0 lo32
    g1[6] = (int)(uint32_t)((st0 >> 32) & 0xFFFFu);               // stride0 hi16
    g1[7] = 0;

    v4i gz = {0, 0, 0, 0};
#if TDM_ARGS6
    v8i gz8 = {0, 0, 0, 0, 0, 0, 0, 0};
    __builtin_amdgcn_tensor_load_to_lds(g0, g1, gz, gz, gz8, 0);
#else
    __builtin_amdgcn_tensor_load_to_lds(g0, g1, gz, gz, 0);
#endif
}
#endif

// ---------------------------------------------------------------------------
// WMMA fragment loaders (CDNA5 16x16x32 bf16 layouts, ISA 7.12.2)
// ---------------------------------------------------------------------------
__device__ __forceinline__ v16bf load_a_frag(const __bf16* base, int ld) {
    const int lane = threadIdx.x & 31;
    const int m    = lane & 15;
    const int koff = (lane >> 4) * 8;
    const __bf16* p = base + m * ld + koff;
    union { v16bf v; uint4 q[2]; } u;
    u.q[0] = *(const uint4*)(p);
    u.q[1] = *(const uint4*)(p + 16);
    return u.v;
}

__device__ __forceinline__ v16bf load_b_frag(const __bf16* base, int ld) {
    const int lane = threadIdx.x & 31;
    const int n    = lane & 15;
    const int koff = (lane >> 4) * 8;
    const __bf16* p = base + n * ld + koff;
    union { v16bf v; uint4 q[2]; } u;
    u.q[0] = *(const uint4*)(p);
    u.q[1] = *(const uint4*)(p + 16);
    return u.v;
}

__device__ __forceinline__ v8f wmma_bf16(v16bf a, v16bf b, v8f c) {
    return __builtin_amdgcn_wmma_f32_16x16x32_bf16(false, a, false, b,
                                                   (short)0, c, false, false);
}

// ---------------------------------------------------------------------------
// fp32 -> bf16 elementwise conversion
// ---------------------------------------------------------------------------
__global__ void cvt_bf16_kernel(const float* __restrict__ in,
                                __bf16* __restrict__ out, int n) {
    int i = blockIdx.x * blockDim.x + threadIdx.x;
    if (i < n) out[i] = (__bf16)in[i];
}

// ---------------------------------------------------------------------------
// LayerNorm over C=1024, one block per row; writes bf16 nx
// ---------------------------------------------------------------------------
__global__ void layernorm_kernel(const float* __restrict__ x,
                                 const float* __restrict__ w,
                                 __bf16* __restrict__ nx) {
    __shared__ float sred[256];
    const int row = blockIdx.x;
    const int tid = threadIdx.x;
    const float* xr = x + (size_t)row * CC;

    float xv[4];
    float s = 0.f;
#pragma unroll
    for (int i = 0; i < 4; ++i) {
        xv[i] = xr[i * 256 + tid];
        s += xv[i];
    }
    sred[tid] = s; __syncthreads();
    for (int o = 128; o > 0; o >>= 1) {
        if (tid < o) sred[tid] += sred[tid + o];
        __syncthreads();
    }
    const float mu = sred[0] * (1.0f / CC);
    __syncthreads();

    float v = 0.f;
#pragma unroll
    for (int i = 0; i < 4; ++i) {
        float d = xv[i] - mu;
        v += d * d;
    }
    sred[tid] = v; __syncthreads();
    for (int o = 128; o > 0; o >>= 1) {
        if (tid < o) sred[tid] += sred[tid + o];
        __syncthreads();
    }
    const float rstd = rsqrtf(sred[0] * (1.0f / CC) + 1e-5f);

    __bf16* nr = nx + (size_t)row * CC;
#pragma unroll
    for (int i = 0; i < 4; ++i) {
        int c = i * 256 + tid;
        nr[c] = (__bf16)((xv[i] - mu) * rstd * w[c]);
    }
}

// ---------------------------------------------------------------------------
// Causal running mean of nx over T (per batch, per channel): the MC@v term
// ---------------------------------------------------------------------------
__global__ void cummean_kernel(const __bf16* __restrict__ nx,
                               float* __restrict__ cm) {
    const int c = blockIdx.x * blockDim.x + threadIdx.x;  // 0..1023
    const int b = blockIdx.y;
    float run = 0.f;
    for (int t = 0; t < TT; ++t) {
        size_t idx = ((size_t)b * TT + t) * CC + c;
        run += (float)nx[idx];
        cm[idx] = run / (float)(t + 1);
    }
}

// ---------------------------------------------------------------------------
// Pre-transpose V per head: vt[b][h][d][t] = nx[b][t][h*DD+d]
// Makes the attention V tile a plain 2D tile -> loadable by TDM.
// ---------------------------------------------------------------------------
__global__ void transpose_v_kernel(const __bf16* __restrict__ nx,
                                   __bf16* __restrict__ vt) {
    int i = blockIdx.x * blockDim.x + threadIdx.x;   // over B*T*C, c fastest
    int c = i & (CC - 1);
    int t = (i >> 10) & (TT - 1);
    int b = i >> 21;
    int h = c >> 6;
    int d = c & 63;
    vt[(((size_t)b * HH + h) * DD + d) * TT + t] = nx[i];
}

// ---------------------------------------------------------------------------
// Generic bf16 WMMA GEMM: out[M,N] = A[M,K] @ W[N,K]^T
//   block 256 thr (8 waves), tile 128x128, BK=32, double-buffered TDM staging
//   (issue slab i+1, s_wait_tensorcnt(1) completes slab i in-order while the
//   next DMA overlaps the WMMA chain).
// ---------------------------------------------------------------------------
template <int MODE>
__global__ __launch_bounds__(256, 1)
void gemm_kernel(const __bf16* __restrict__ A,
                 const __bf16* __restrict__ Wt,
                 void* __restrict__ outp,
                 int M, int N, int K,
                 const float* __restrict__ xg,
                 const float* __restrict__ yg,
                 const __bf16* __restrict__ nxg,
                 const float* __restrict__ cmg,
                 const float* pAlpha, const float* pBeta,
                 const float* pGamma, const float* pBSA,
                 const float* pBFF) {
    __shared__ __align__(16) __bf16 As[2][128][40];
    __shared__ __align__(16) __bf16 Bs[2][128][40];

    const int tid  = threadIdx.x;
    const int wid  = tid >> 5;
    const int lane = tid & 31;
    const int wm   = wid & 3;        // 4 M-strips of 32
    const int wn   = wid >> 2;       // 2 N-strips of 64
    const int bn0  = blockIdx.x * 128;
    const int bm0  = blockIdx.y * 128;

    v8f acc[2][4] = {};

#if HAVE_TDM
    // prologue: issue slab 0
    if (wid == 0)
        tdm_load_2d(A + (size_t)bm0 * K, (uint32_t)(uintptr_t)&As[0][0][0],
                    128, 32, (uint64_t)K, 128, 3, 3);
    if (wid == 1)
        tdm_load_2d(Wt + (size_t)bn0 * K, (uint32_t)(uintptr_t)&Bs[0][0][0],
                    128, 32, (uint64_t)K, 128, 3, 3);
#endif

    int ib = 0;
    for (int kb = 0; kb < K; kb += 32, ib ^= 1) {
#if HAVE_TDM
        if (kb + 32 < K) {
            if (wid == 0)
                tdm_load_2d(A + (size_t)bm0 * K + kb + 32,
                            (uint32_t)(uintptr_t)&As[ib ^ 1][0][0],
                            128, 32, (uint64_t)K, 128, 3, 3);
            if (wid == 1)
                tdm_load_2d(Wt + (size_t)bn0 * K + kb + 32,
                            (uint32_t)(uintptr_t)&Bs[ib ^ 1][0][0],
                            128, 32, (uint64_t)K, 128, 3, 3);
            if (wid < 2) __builtin_amdgcn_s_wait_tensorcnt(1);
        } else {
            if (wid < 2) __builtin_amdgcn_s_wait_tensorcnt(0);
        }
        __syncthreads();   // current slab visible to all waves
#else
        __syncthreads();
#pragma unroll
        for (int it = 0; it < 2; ++it) {
            int ci  = tid + 256 * it;          // 0..511
            int row = ci >> 2;
            int c8  = (ci & 3) * 8;
            *(uint4*)(&As[0][row][c8]) =
                *(const uint4*)(A + (size_t)(bm0 + row) * K + kb + c8);
            *(uint4*)(&Bs[0][row][c8]) =
                *(const uint4*)(Wt + (size_t)(bn0 + row) * K + kb + c8);
        }
        __syncthreads();
#endif
        const int rb = HAVE_TDM ? ib : 0;
        v16bf af[2], bf[4];
#pragma unroll
        for (int i = 0; i < 2; ++i)
            af[i] = load_a_frag(&As[rb][32 * wm + 16 * i][0], 40);
#pragma unroll
        for (int j = 0; j < 4; ++j)
            bf[j] = load_b_frag(&Bs[rb][64 * wn + 16 * j][0], 40);
#pragma unroll
        for (int i = 0; i < 2; ++i)
#pragma unroll
            for (int j = 0; j < 4; ++j)
                acc[i][j] = wmma_bf16(af[i], bf[j], acc[i][j]);
        __syncthreads();   // reads done before this buffer is refilled
    }

    const int half = lane >> 4;
    const int nl   = lane & 15;

    float alpha = 0.f, beta = 0.f, gamma = 0.f, bSA = 0.f, bFF = 0.f;
    if (MODE == 2) {
        alpha = *pAlpha; beta = *pBeta; gamma = *pGamma;
        bSA = *pBSA; bFF = *pBFF;
    }

#pragma unroll
    for (int i = 0; i < 2; ++i) {
#pragma unroll
        for (int j = 0; j < 4; ++j) {
#pragma unroll
            for (int r = 0; r < 8; ++r) {
                int m = bm0 + 32 * wm + 16 * i + r + 8 * half;
                int n = bn0 + 64 * wn + 16 * j + nl;
                float v = acc[i][j][r];
                if (MODE == 0) {
                    ((__bf16*)outp)[(size_t)m * N + n] = (__bf16)v;
                } else if (MODE == 1) {
                    float g = 0.5f * v * (1.0f + erff(v * 0.70710678118654752f));
                    ((__bf16*)outp)[(size_t)m * N + n] = (__bf16)g;
                } else {
                    size_t idx = (size_t)m * N + n;   // N == CC here
                    float yv = beta * yg[idx] + alpha * (float)nxg[idx]
                               - gamma * cmg[idx];
                    ((float*)outp)[idx] = xg[idx] + bSA * yv + bFF * v;
                }
            }
        }
    }
}

// ---------------------------------------------------------------------------
// Flash attention (causal), WMMA for Q@K^T and P@V.
//   grid: (T/64, B*H); block 128 thr (4 waves); each wave owns 16 query rows.
//   K tiles (wave0) and pre-transposed V tiles (wave1) staged by TDM with
//   double buffering: each wave's private TENSORcnt + in-order completion
//   lets s_wait_tensorcnt(1) retire the current tile while the next streams.
// ---------------------------------------------------------------------------
__global__ __launch_bounds__(128, 1)
void attn_kernel(const __bf16* __restrict__ qk,
                 const __bf16* __restrict__ vt,
                 float* __restrict__ y) {
    __shared__ __align__(16) __bf16 Qs[64][72];
    __shared__ __align__(16) __bf16 Ks[2][64][72];
    __shared__ __align__(16) __bf16 Vts[2][64][72];   // [d][key]
    __shared__ __align__(16) __bf16 Ps[4][16][72];    // per-wave P tile

    const int tid  = threadIdx.x;
    const int wid  = tid >> 5;
    const int lane = tid & 31;
    const int half = lane >> 4;
    const int nl   = lane & 15;

    const int qt = blockIdx.x;            // query tile (64 rows)
    const int bh = blockIdx.y;
    const int b  = bh >> 4;
    const int h  = bh & 15;
    const int qbase = qt * 64;

    const __bf16* kbase_p = qk + (size_t)b * TT * (2 * CC) + CC + h * DD;
    const __bf16* vbase_p = vt + (((size_t)b * HH + h) * DD) * TT;

#if HAVE_TDM
    if (wid == 0) {
        tdm_load_2d(qk + ((size_t)(b * TT + qbase)) * (2 * CC) + h * DD,
                    (uint32_t)(uintptr_t)&Qs[0][0], 64, 64,
                    (uint64_t)(2 * CC), 64, 4, 3);
        tdm_load_2d(kbase_p, (uint32_t)(uintptr_t)&Ks[0][0][0], 64, 64,
                    (uint64_t)(2 * CC), 64, 4, 3);
    }
    if (wid == 1)
        tdm_load_2d(vbase_p, (uint32_t)(uintptr_t)&Vts[0][0][0], 64, 64,
                    (uint64_t)TT, 64, 4, 3);
    if (wid < 2) __builtin_amdgcn_s_wait_tensorcnt(0);
#else
#pragma unroll
    for (int it = 0; it < 4; ++it) {
        int ci  = tid + 128 * it;
        int row = ci >> 3;
        int c8  = (ci & 7) * 8;
        *(uint4*)(&Qs[row][c8]) =
            *(const uint4*)(qk + ((size_t)(b * TT + qbase + row)) * (2 * CC)
                               + h * DD + c8);
    }
#endif
    __syncthreads();

    // per-wave Q A-fragments (K-dim = 64 -> 2 fragments), kept in registers
    v16bf qa[2];
#pragma unroll
    for (int ks = 0; ks < 2; ++ks)
        qa[ks] = load_a_frag(&Qs[16 * wid][32 * ks], 72);

    v8f   o[4] = {};
    float rmax[8], rsum[8];
#pragma unroll
    for (int r = 0; r < 8; ++r) { rmax[r] = -1e30f; rsum[r] = 0.f; }

    for (int kt = 0; kt <= qt; ++kt) {
        const int kbase = kt * 64;
        const int ib = HAVE_TDM ? (kt & 1) : 0;
#if HAVE_TDM
        if (kt < qt) {   // issue next tiles, then retire current (in-order)
            if (wid == 0)
                tdm_load_2d(kbase_p + (size_t)(kbase + 64) * (2 * CC),
                            (uint32_t)(uintptr_t)&Ks[ib ^ 1][0][0], 64, 64,
                            (uint64_t)(2 * CC), 64, 4, 3);
            if (wid == 1)
                tdm_load_2d(vbase_p + (kbase + 64),
                            (uint32_t)(uintptr_t)&Vts[ib ^ 1][0][0], 64, 64,
                            (uint64_t)TT, 64, 4, 3);
            if (wid < 2) __builtin_amdgcn_s_wait_tensorcnt(1);
        } else {
            if (wid < 2) __builtin_amdgcn_s_wait_tensorcnt(0);
        }
        __syncthreads();
#else
        __syncthreads();
#pragma unroll
        for (int it = 0; it < 4; ++it) {
            int ci  = tid + 128 * it;
            int row = ci >> 3;
            int c8  = (ci & 7) * 8;
            *(uint4*)(&Ks[0][row][c8]) =
                *(const uint4*)(kbase_p + (size_t)(kbase + row) * (2 * CC) + c8);
            *(uint4*)(&Vts[0][row][c8]) =
                *(const uint4*)(vbase_p + (size_t)row * TT + kbase + c8);
        }
        __syncthreads();
#endif

        // S = Q @ K^T * d^-0.5 : batch B-fragments, then back-to-back WMMAs
        v8f sacc[4] = {};
#pragma unroll
        for (int ks = 0; ks < 2; ++ks) {
            v16bf kbf[4];
#pragma unroll
            for (int j = 0; j < 4; ++j)
                kbf[j] = load_b_frag(&Ks[ib][16 * j][32 * ks], 72);
#pragma unroll
            for (int j = 0; j < 4; ++j)
                sacc[j] = wmma_bf16(qa[ks], kbf[j], sacc[j]);
        }
#pragma unroll
        for (int j = 0; j < 4; ++j)
#pragma unroll
            for (int r = 0; r < 8; ++r)
                sacc[j][r] *= 0.125f;                 // 1/sqrt(64)

        if (kt == qt) {                               // causal mask on diagonal
#pragma unroll
            for (int j = 0; j < 4; ++j) {
                int key = kbase + 16 * j + nl;
#pragma unroll
                for (int r = 0; r < 8; ++r) {
                    int q = qbase + wid * 16 + r + 8 * half;
                    if (key > q) sacc[j][r] = -1e30f;
                }
            }
        }

        // online softmax update
#pragma unroll
        for (int r = 0; r < 8; ++r) {
            float tmax = sacc[0][r];
#pragma unroll
            for (int j = 1; j < 4; ++j) tmax = fmaxf(tmax, sacc[j][r]);
#pragma unroll
            for (int msk = 1; msk < 16; msk <<= 1)
                tmax = fmaxf(tmax, __shfl_xor(tmax, msk, 32));
            float nm = fmaxf(rmax[r], tmax);
            float sc = __expf(rmax[r] - nm);
            rmax[r] = nm;
            float rs = 0.f;
#pragma unroll
            for (int j = 0; j < 4; ++j) {
                sacc[j][r] = __expf(sacc[j][r] - nm);
                rs += sacc[j][r];
            }
#pragma unroll
            for (int msk = 1; msk < 16; msk <<= 1)
                rs += __shfl_xor(rs, msk, 32);
            rsum[r] = rsum[r] * sc + rs;
#pragma unroll
            for (int j = 0; j < 4; ++j) o[j][r] *= sc;
        }

        // P: C-layout -> LDS (row major) -> reload in A-layout
#pragma unroll
        for (int j = 0; j < 4; ++j)
#pragma unroll
            for (int r = 0; r < 8; ++r)
                Ps[wid][r + 8 * half][16 * j + nl] = (__bf16)sacc[j][r];
        __syncthreads();

        // O += P @ V
#pragma unroll
        for (int ks = 0; ks < 2; ++ks) {
            v16bf pa = load_a_frag(&Ps[wid][0][32 * ks], 72);
            v16bf vbf[4];
#pragma unroll
            for (int j = 0; j < 4; ++j)
                vbf[j] = load_b_frag(&Vts[ib][16 * j][32 * ks], 72);
#pragma unroll
            for (int j = 0; j < 4; ++j)
                o[j] = wmma_bf16(pa, vbf[j], o[j]);
        }
        __syncthreads();   // buffer reads done before refill
    }

    // normalize and write y
#pragma unroll
    for (int r = 0; r < 8; ++r) {
        float inv = 1.0f / rsum[r];
        int t = qbase + wid * 16 + r + 8 * half;
#pragma unroll
        for (int j = 0; j < 4; ++j)
            y[((size_t)(b * TT + t)) * CC + h * DD + 16 * j + nl] = o[j][r] * inv;
    }
}

// ---------------------------------------------------------------------------
// Launch
// ---------------------------------------------------------------------------
extern "C" void kernel_launch(void* const* d_in, const int* in_sizes, int n_in,
                              void* d_out, int out_size, void* d_ws, size_t ws_size,
                              hipStream_t stream) {
    const float* x      = (const float*)d_in[0];
    const float* ln_w   = (const float*)d_in[1];
    const float* W_attn = (const float*)d_in[2];
    const float* W_fc   = (const float*)d_in[3];
    const float* W_proj = (const float*)d_in[4];
    const float* alpha  = (const float*)d_in[5];
    const float* beta   = (const float*)d_in[6];
    const float* gamma  = (const float*)d_in[7];
    const float* bSA    = (const float*)d_in[8];
    const float* bFF    = (const float*)d_in[9];
    float* out = (float*)d_out;

    // workspace layout
    char* ws = (char*)d_ws;
    size_t off = 0;
    __bf16* nx_bf = (__bf16*)(ws + off); off += (size_t)MM * CC * 2;        // 8 MB
    __bf16* qk_bf = (__bf16*)(ws + off); off += (size_t)MM * 2 * CC * 2;    // 16 MB
    __bf16* h_bf  = (__bf16*)(ws + off); off += (size_t)MM * NFF * 2;       // 32 MB
    float*  y_f   = (float*) (ws + off); off += (size_t)MM * CC * 4;        // 16 MB
    float*  cm_f  = (float*) (ws + off); off += (size_t)MM * CC * 4;        // 16 MB
    __bf16* vt_bf = (__bf16*)(ws + off); off += (size_t)MM * CC * 2;        // 8 MB
    __bf16* Wa_bf = (__bf16*)(ws + off); off += (size_t)2 * CC * CC * 2;    // 4 MB
    __bf16* Wf_bf = (__bf16*)(ws + off); off += (size_t)NFF * CC * 2;       // 8 MB
    __bf16* Wp_bf = (__bf16*)(ws + off); off += (size_t)CC * NFF * 2;       // 8 MB

    // 1) weights -> bf16
    cvt_bf16_kernel<<<(2 * CC * CC + 255) / 256, 256, 0, stream>>>(W_attn, Wa_bf, 2 * CC * CC);
    cvt_bf16_kernel<<<(NFF * CC + 255) / 256, 256, 0, stream>>>(W_fc, Wf_bf, NFF * CC);
    cvt_bf16_kernel<<<(CC * NFF + 255) / 256, 256, 0, stream>>>(W_proj, Wp_bf, CC * NFF);

    // 2) layernorm
    layernorm_kernel<<<MM, 256, 0, stream>>>(x, ln_w, nx_bf);

    // 3) causal running mean of v (= nx); 4) per-head V transpose
    cummean_kernel<<<dim3(CC / 256, BB), 256, 0, stream>>>(nx_bf, cm_f);
    transpose_v_kernel<<<(MM * CC) / 256, 256, 0, stream>>>(nx_bf, vt_bf);

    // 5) qk projection: [4096,1024] @ [2048,1024]^T -> bf16 [4096,2048]
    gemm_kernel<0><<<dim3(2 * CC / 128, MM / 128), 256, 0, stream>>>(
        nx_bf, Wa_bf, qk_bf, MM, 2 * CC, CC,
        nullptr, nullptr, nullptr, nullptr,
        nullptr, nullptr, nullptr, nullptr, nullptr);

    // 6) flash attention -> y (att@v only; alpha/gamma folded later)
    attn_kernel<<<dim3(TT / 64, BB * HH), 128, 0, stream>>>(qk_bf, vt_bf, y_f);

    // 7) fc + exact GELU: [4096,1024] @ [4096,1024]^T -> bf16 [4096,4096]
    gemm_kernel<1><<<dim3(NFF / 128, MM / 128), 256, 0, stream>>>(
        nx_bf, Wf_bf, h_bf, MM, NFF, CC,
        nullptr, nullptr, nullptr, nullptr,
        nullptr, nullptr, nullptr, nullptr, nullptr);

    // 8) proj + fused residual epilogue:
    //    out = x + bSA*(beta*y + alpha*nx - gamma*cm) + bFF*(h @ Wp^T)
    gemm_kernel<2><<<dim3(CC / 128, MM / 128), 256, 0, stream>>>(
        h_bf, Wp_bf, out, MM, CC, NFF,
        x, y_f, nx_bf, cm_f,
        alpha, beta, gamma, bSA, bFF);
}
